// TransformerCustomEncoder_62405874811685
// MI455X (gfx1250) — matrix operand order repaired and checked by reading, not compiled, not run on hardware
//
#include <hip/hip_runtime.h>
#include <cmath>

// ---------------- model constants ----------------
#define NB   16      // batch
#define NTOK 280     // text(180) + visn(100), text rows first
#define NTXT 180
#define NVIS 100
#define BD   512     // model dim
#define FD   2048    // ffn dim
#define NH   8       // heads (dh = 64)

typedef __attribute__((ext_vector_type(16))) __bf16 v16bf;
typedef __attribute__((ext_vector_type(8)))  __bf16 v8bf;
typedef __attribute__((ext_vector_type(8)))  float  v8f;

// pack two f32 -> two bf16 (RNE) in one u32
__device__ __forceinline__ unsigned bf16pk(float lo, float hi) {
  unsigned ul = __builtin_bit_cast(unsigned, lo);
  unsigned uh = __builtin_bit_cast(unsigned, hi);
  ul = (ul + 0x7FFFu + ((ul >> 16) & 1u)) >> 16;
  uh = (uh + 0x7FFFu + ((uh >> 16) & 1u)) & 0xFFFF0000u;
  return ul | uh;
}

__device__ __forceinline__ float apply_act(float x, int act) {
  if (act == 1) return fmaxf(x, 0.f);                                     // relu
  if (act == 2) return 0.5f * x * (1.f + erff(x * 0.70710678118654752f)); // exact gelu
  return x;
}

// =====================================================================
// WMMA GEMM:  C[rowmap(m), 0..N) = act( (A[rowmap(m),:] (+A2)) @ W + bias )
// A: f32 [*, Kd] row-major via rowmap, W: f32 [Kd, N] row-major.
// rowmap(m) = (m / rowCount) * rowPitch + rowStart + (m % rowCount)
// Assumptions (hold for every launch here): M % 64 == 0, N % 128 == 0,
// Kd % 32 == 0.
// 64(M) x 128(N) tile / block, 8 waves (4Mx2N), each wave = 16x64 output
// (4 WMMA 16x16x32 bf16 per K-step, A fragment reused 4x), fp32 accum.
// Double-buffered LDS, software-pipelined global->LDS staging.
// =====================================================================
#define TBM 64
#define TBN 128
#define TBK 32
#define LDT 40    // padded K-stride in LDS (bf16 elems) = 80B rows (16B aligned)

template <bool HAS_A2, int ACT>
__global__ __launch_bounds__(256)
void gemm_kernel(const float* __restrict__ A, const float* __restrict__ A2,
                 const float* __restrict__ W, const float* __restrict__ bias,
                 float* __restrict__ C, int M, int N, int Kd,
                 int rowStart, int rowCount, int rowPitch)
{
  (void)M;
  __shared__ __align__(16) __bf16 As[2][TBM * LDT];   //  2 x 5 KB
  __shared__ __align__(16) __bf16 Bs[2][TBN * LDT];   //  2 x 10 KB

  const int tid  = threadIdx.x;
  const int lane = tid & 31;
  const int wave = tid >> 5;
  const int wr   = wave >> 1;          // 0..3 : M sub-tile (16 rows)
  const int wc   = wave & 1;           // 0..1 : N sub-tile (64 cols)
  const int m0   = blockIdx.y * TBM;
  const int n0   = blockIdx.x * TBN;
  const int lm   = lane & 15;
  const int kbA  = (lane >> 4) * 16;   // A frag: contiguous K=kbA..kbA+15
  const int kbB  = (lane >> 4) * 16;   // B frag: contiguous K=kbB..kbB+15

  // ---- per-thread staging descriptors (hoisted out of the K loop) ----
  // A tile: 64 rows x 32 cols f32 = 512 float4, 2 per thread
  const float* aptr[2]; const float* a2ptr[2]; int aoff[2];
  #pragma unroll
  for (int i = 0; i < 2; ++i) {
    int idx = tid + i * 256;
    int r = idx >> 3;                 // row in tile
    int c = (idx & 7) * 4;            // col (k) in tile
    int gm = m0 + r;
    int row = (gm / rowCount) * rowPitch + rowStart + (gm % rowCount);
    aptr[i]  = A + (size_t)row * Kd + c;
    a2ptr[i] = HAS_A2 ? (A2 + (size_t)row * Kd + c) : nullptr;
    aoff[i]  = r * LDT + c;
  }
  // B tile: 32(K) x 128(N) f32, staged transposed; 4 groups of 4-K per thread
  const float* wptr[4]; int boff[4];
  #pragma unroll
  for (int i = 0; i < 4; ++i) {
    int idx = tid + i * 256;
    int n  = idx & 127;
    int cg = idx >> 7;                // 0..7 -> k base cg*4
    wptr[i] = W + (size_t)(cg * 4) * N + n0 + n;
    boff[i] = n * LDT + cg * 4;
  }

  float4 va[2];
  float  vb[4][4];

  auto load_tiles = [&](int k0) {
    #pragma unroll
    for (int i = 0; i < 2; ++i) {
      float4 v = *(const float4*)(aptr[i] + k0);
      if (HAS_A2) {
        float4 u = *(const float4*)(a2ptr[i] + k0);
        v.x += u.x; v.y += u.y; v.z += u.z; v.w += u.w;
      }
      va[i] = v;
    }
    #pragma unroll
    for (int i = 0; i < 4; ++i)
      #pragma unroll
      for (int j = 0; j < 4; ++j)
        vb[i][j] = wptr[i][(size_t)(k0 + j) * N];   // coalesced across lanes (n)
  };
  auto store_tiles = [&](int buf) {
    #pragma unroll
    for (int i = 0; i < 2; ++i) {
      uint2 p; p.x = bf16pk(va[i].x, va[i].y); p.y = bf16pk(va[i].z, va[i].w);
      *reinterpret_cast<uint2*>(&As[buf][aoff[i]]) = p;   // ds_store_b64
    }
    #pragma unroll
    for (int i = 0; i < 4; ++i) {
      uint2 p; p.x = bf16pk(vb[i][0], vb[i][1]); p.y = bf16pk(vb[i][2], vb[i][3]);
      *reinterpret_cast<uint2*>(&Bs[buf][boff[i]]) = p;
    }
  };

  v8f c0 = {}, c1 = {}, c2 = {}, c3 = {};

  // ---- pipelined main loop ----
  load_tiles(0);
  store_tiles(0);
  __syncthreads();

  const int T = Kd / TBK;
  for (int t = 0; t < T; ++t) {
    const int cb = t & 1;
    if (t + 1 < T) {
      load_tiles((t + 1) * TBK);                       // overlap with WMMAs
      if (t + 2 < T)                                   // global_prefetch_b8
        __builtin_prefetch(wptr[0] + (size_t)(t + 2) * TBK * N, 0, 1);
    }

    union F16 { v16bf v; v8bf h[2]; };
    F16 af, b0f, b1f, b2f, b3f;
    const int ra = (wr * 16 + lm) * LDT;
    af.h[0] = *(const v8bf*)&As[cb][ra + kbA];
    af.h[1] = *(const v8bf*)&As[cb][ra + kbA + 8];
    const int rb = (wc * 64 + lm) * LDT + kbB;
    b0f.h[0] = *(const v8bf*)&Bs[cb][rb];
    b0f.h[1] = *(const v8bf*)&Bs[cb][rb + 8];
    b1f.h[0] = *(const v8bf*)&Bs[cb][rb + 16 * LDT];
    b1f.h[1] = *(const v8bf*)&Bs[cb][rb + 16 * LDT + 8];
    b2f.h[0] = *(const v8bf*)&Bs[cb][rb + 32 * LDT];
    b2f.h[1] = *(const v8bf*)&Bs[cb][rb + 32 * LDT + 8];
    b3f.h[0] = *(const v8bf*)&Bs[cb][rb + 48 * LDT];
    b3f.h[1] = *(const v8bf*)&Bs[cb][rb + 48 * LDT + 8];

    c0 = __builtin_amdgcn_wmma_f32_16x16x32_bf16(false, af.v, false, b0f.v, (short)0, c0, false, false);
    c1 = __builtin_amdgcn_wmma_f32_16x16x32_bf16(false, af.v, false, b1f.v, (short)0, c1, false, false);
    c2 = __builtin_amdgcn_wmma_f32_16x16x32_bf16(false, af.v, false, b2f.v, (short)0, c2, false, false);
    c3 = __builtin_amdgcn_wmma_f32_16x16x32_bf16(false, af.v, false, b3f.v, (short)0, c3, false, false);

    if (t + 1 < T) store_tiles(cb ^ 1);
    __syncthreads();
  }

  // ---- epilogue: bias + activation + store (M % 64 == 0 -> no guards) ----
  const int mrow = m0 + wr * 16 + (lane >> 4) * 8;
  const int nc   = n0 + wc * 64 + lm;
  #pragma unroll
  for (int r = 0; r < 8; ++r) {
    int gm  = mrow + r;
    int row = (gm / rowCount) * rowPitch + rowStart + (gm % rowCount);
    float* crow = C + (size_t)row * N;
    crow[nc]      = apply_act(c0[r] + bias[nc],      ACT);
    crow[nc + 16] = apply_act(c1[r] + bias[nc + 16], ACT);
    crow[nc + 32] = apply_act(c2[r] + bias[nc + 32], ACT);
    crow[nc + 48] = apply_act(c3[r] + bias[nc + 48], ACT);
  }
}

// =====================================================================
// Residual add + LayerNorm over D=512 (one block / token)
// =====================================================================
__global__ __launch_bounds__(256)
void add_layernorm(const float* __restrict__ X, const float* __restrict__ Y,
                   const float* __restrict__ g, const float* __restrict__ b,
                   float* __restrict__ O, int rowStart, int rowCount, int rowPitch)
{
  int blk = blockIdx.x;
  int row = (blk / rowCount) * rowPitch + rowStart + (blk % rowCount);
  const float* x = X + (size_t)row * BD;
  const float* y = Y + (size_t)row * BD;
  float* o = O + (size_t)row * BD;
  int tid = threadIdx.x;
  float v0 = x[tid] + y[tid];
  float v1 = x[tid + 256] + y[tid + 256];
  __shared__ float red[256];
  red[tid] = v0 + v1;
  __syncthreads();
  for (int s = 128; s > 0; s >>= 1) { if (tid < s) red[tid] += red[tid + s]; __syncthreads(); }
  float mean = red[0] * (1.f / 512.f);
  __syncthreads();
  float d0 = v0 - mean, d1 = v1 - mean;
  red[tid] = d0 * d0 + d1 * d1;
  __syncthreads();
  for (int s = 128; s > 0; s >>= 1) { if (tid < s) red[tid] += red[tid + s]; __syncthreads(); }
  float rstd = rsqrtf(red[0] * (1.f / 512.f) + 1e-12f);
  o[tid]       = d0 * rstd * g[tid]       + b[tid];
  o[tid + 256] = d1 * rstd * g[tid + 256] + b[tid + 256];
}

// =====================================================================
// Attention: softmax((Q Kt)/8 + rpe + mask) V, one block per (b, h, q-row)
// Q/K/V/O laid out [B*280, 512]; head h = columns h*64..h*64+63
// =====================================================================
__global__ __launch_bounds__(128)
void attention(const float* __restrict__ Q, const float* __restrict__ K,
               const float* __restrict__ V, const float* __restrict__ G,
               const float* __restrict__ mask, int mlen,
               float* __restrict__ O,
               int qs, int Lq, int ks, int Lk, int gq0, int gk0)
{
  int qi = blockIdx.x, h = blockIdx.y, b = blockIdx.z;
  (void)Lq;
  __shared__ float qv[64];
  __shared__ float sc[288];
  __shared__ float red[128];
  int tid = threadIdx.x;
  size_t qrow = (size_t)(b * NTOK + qs + qi);
  int ho = h * 64;
  if (tid < 64) qv[tid] = Q[qrow * BD + ho + tid];
  __syncthreads();

  float lmax = -3.0e38f;
  for (int kk = tid; kk < Lk; kk += 128) {
    const float* kp = K + ((size_t)(b * NTOK + ks + kk)) * BD + ho;
    float acc = 0.f;
    #pragma unroll 8
    for (int d = 0; d < 64; ++d) acc += qv[d] * kp[d];
    float s = acc * 0.125f
            + G[(((size_t)(b * NH + h)) * NTOK + (gq0 + qi)) * NTOK + (gk0 + kk)];
    if (mask) s += mask[b * mlen + kk];
    sc[kk] = s;
    lmax = fmaxf(lmax, s);
  }
  red[tid] = lmax; __syncthreads();
  for (int s = 64; s > 0; s >>= 1) { if (tid < s) red[tid] = fmaxf(red[tid], red[tid + s]); __syncthreads(); }
  float mx = red[0]; __syncthreads();
  float lsum = 0.f;
  for (int kk = tid; kk < Lk; kk += 128) { float e = expf(sc[kk] - mx); sc[kk] = e; lsum += e; }
  red[tid] = lsum; __syncthreads();
  for (int s = 64; s > 0; s >>= 1) { if (tid < s) red[tid] += red[tid + s]; __syncthreads(); }
  float rinv = 1.f / red[0];
  __syncthreads();
  if (tid < 64) {
    float acc = 0.f;
    for (int kk = 0; kk < Lk; ++kk)
      acc += sc[kk] * V[((size_t)(b * NTOK + ks + kk)) * BD + ho + tid];
    O[qrow * BD + ho + tid] = acc * rinv;
  }
}

// =====================================================================
// Box relation embedding -> g[b, h, m, n] = relu(emb64 . fc_g_w[:,h] + fc_g_b[h])
// =====================================================================
__global__ void rel_embed(const float* __restrict__ boxes,
                          const float* __restrict__ fcw, const float* __restrict__ fcb,
                          float* __restrict__ G)
{
  int idx = blockIdx.x * blockDim.x + threadIdx.x;
  if (idx >= NB * NTOK * NTOK) return;
  int n = idx % NTOK;
  int m = (idx / NTOK) % NTOK;
  int b = idx / (NTOK * NTOK);
  const float* bm = boxes + ((size_t)b * NTOK + m) * 4;
  const float* bn = boxes + ((size_t)b * NTOK + n) * 4;
  float cxm = (bm[0] + bm[2]) * 0.5f, cym = (bm[1] + bm[3]) * 0.5f;
  float wm = bm[2] - bm[0] + 1.f,     hm = bm[3] - bm[1] + 1.f;
  float cxn = (bn[0] + bn[2]) * 0.5f, cyn = (bn[1] + bn[3]) * 0.5f;
  float wn = bn[2] - bn[0] + 1.f,     hn = bn[3] - bn[1] + 1.f;
  float pos[4];
  pos[0] = logf(fmaxf(fabsf(cxm - cxn) / wm, 0.001f));
  pos[1] = logf(fmaxf(fabsf(cym - cyn) / hm, 0.001f));
  pos[2] = logf(wm / wn);
  pos[3] = logf(hm / hn);
  float acc[NH];
  #pragma unroll
  for (int h = 0; h < NH; ++h) acc[h] = fcb[h];
  for (int j = 0; j < 8; ++j) {
    float dm  = expf((float)j * 0.86346940987f);  // 1000^(j/8)
    float inv = 100.f / dm;
    #pragma unroll
    for (int p = 0; p < 4; ++p) {
      float dv = pos[p] * inv;
      float sn = sinf(dv), cs = cosf(dv);
      int ksn = p * 16 + j, kcs = p * 16 + 8 + j;
      #pragma unroll
      for (int h = 0; h < NH; ++h)
        acc[h] += sn * fcw[ksn * NH + h] + cs * fcw[kcs * NH + h];
    }
  }
  size_t base = ((size_t)b * NH) * NTOK * NTOK + (size_t)m * NTOK + n;
  #pragma unroll
  for (int h = 0; h < NH; ++h)
    G[base + (size_t)h * NTOK * NTOK] = fmaxf(acc[h], 0.f);
}

// absolute position embeddings: rows <180 text weights, >=180 region weights
__global__ __launch_bounds__(256)
void ape_kernel(const float* __restrict__ boxes,
                const float* __restrict__ tw, const float* __restrict__ tb,
                const float* __restrict__ rw, const float* __restrict__ rb,
                float* __restrict__ cape)
{
  int row = blockIdx.x;
  int b = row / NTOK, r = row % NTOK;
  const float* bx = boxes + ((size_t)b * NTOK + r) * 4;
  const float* Wp = (r < NTXT) ? tw : rw;
  const float* Bp = (r < NTXT) ? tb : rb;
  float x0 = bx[0], x1 = bx[1], x2 = bx[2], x3 = bx[3];
  float* o = cape + (size_t)row * BD;
  for (int d = threadIdx.x; d < BD; d += 256)
    o[d] = x0 * Wp[d] + x1 * Wp[BD + d] + x2 * Wp[2 * BD + d] + x3 * Wp[3 * BD + d] + Bp[d];
}

__global__ void build_mask(const float* __restrict__ vm, float* __restrict__ cm) {
  int b = blockIdx.x, j = threadIdx.x;   // replicate reference cat_mask = [vmask(100), zeros(180)]
  cm[b * NTOK + j] = (j < NVIS) ? vm[b * NVIS + j] : 0.f;
}

__global__ __launch_bounds__(256)
void init_x(const float* __restrict__ visn, const float* __restrict__ text,
            float* __restrict__ X) {
  int row = blockIdx.x;
  int b = row / NTOK, r = row % NTOK;
  const float* src = (r < NTXT) ? text + ((size_t)b * NTXT + r) * BD
                                : visn + ((size_t)b * NVIS + (r - NTXT)) * BD;
  float* dst = X + (size_t)row * BD;
  for (int d = threadIdx.x; d < BD; d += 256) dst[d] = src[d];
}

__global__ __launch_bounds__(256)
void write_out(const float* __restrict__ X, float* __restrict__ out) {
  int row = blockIdx.x;
  int b = row / NTOK, i = row % NTOK;
  int sr = (i < NVIS) ? (NTXT + i) : (i - NVIS);   // output = [visn, text]
  const float* s = X + ((size_t)b * NTOK + sr) * BD;
  float* d = out + (size_t)row * BD;
  for (int t = threadIdx.x; t < BD; t += 256) d[t] = s[t];
}

// =====================================================================
// Host orchestration
// =====================================================================
extern "C" void kernel_launch(void* const* d_in, const int* in_sizes, int n_in,
                              void* d_out, int out_size, void* d_ws, size_t ws_size,
                              hipStream_t stream)
{
  (void)in_sizes; (void)n_in; (void)out_size; (void)ws_size;
  auto F = [&](int i) -> const float* { return (const float*)d_in[i]; };
  const float* visn  = F(0);
  const float* text  = F(1);
  const float* vmask = F(2);
  const float* boxes = F(3);
  const float* rbw = F(4); const float* rbb = F(5);
  const float* tbw = F(6); const float* tbb = F(7);
  const float* fcw = F(8); const float* fcb = F(9);
  // layer params: 50 tensors/layer. modules: self=0 cross=10 mix=20,
  // visn_self_ffn=30 text_self_ffn=36 visn_cross_ffn=42 text_cross_ffn=48
  auto Pp = [&](int l, int mo, int f) -> const float* {
    return (const float*)d_in[10 + l * 50 + mo + f];
  };

  const size_t tokD = (size_t)NB * NTOK * BD;   // 2,293,760 floats
  float* ws = (float*)d_ws;
  size_t off = 0;
  auto alloc = [&](size_t n) { float* p = ws + off; off += n; return p; };
  float* G    = alloc((size_t)NB * NH * NTOK * NTOK);  // 40 MB
  float* CAPE = alloc(tokD);
  float* X    = alloc(tokD);
  float* X2   = alloc(tokD);
  float* Y    = alloc(tokD);
  float* Qb   = alloc(tokD);
  float* Kb   = alloc(tokD);
  float* Vb   = alloc(tokD);
  float* Cb   = alloc(tokD);
  float* Hh   = Qb;                  // FFN hidden [B*280, 2048] aliases Qb..Cb (4*tokD)
  float* CM   = alloc((size_t)NB * NTOK);

  rel_embed<<<(NB * NTOK * NTOK + 255) / 256, 256, 0, stream>>>(boxes, fcw, fcb, G);
  ape_kernel<<<NB * NTOK, 256, 0, stream>>>(boxes, tbw, tbb, rbw, rbb, CAPE);
  build_mask<<<NB, NTOK, 0, stream>>>(vmask, CM);
  init_x<<<NB * NTOK, 256, 0, stream>>>(visn, text, X);

  float* cur = X;
  float* nxt = X2;
  auto swapbuf = [&]() { float* t = cur; cur = nxt; nxt = t; };

  auto gemm = [&](const float* A, const float* A2, const float* Wt, const float* bias,
                  float* Cout, int M, int N, int Kd, int rowStart, int rowCount, int act) {
    dim3 grid(N / TBN, M / TBM);
    if (A2)
      gemm_kernel<true, 0><<<grid, 256, 0, stream>>>(A, A2, Wt, bias, Cout, M, N, Kd,
                                                     rowStart, rowCount, NTOK);
    else if (act == 2)
      gemm_kernel<false, 2><<<grid, 256, 0, stream>>>(A, nullptr, Wt, bias, Cout, M, N, Kd,
                                                      rowStart, rowCount, NTOK);
    else
      gemm_kernel<false, 0><<<grid, 256, 0, stream>>>(A, nullptr, Wt, bias, Cout, M, N, Kd,
                                                      rowStart, rowCount, NTOK);
  };

  struct AttnRange { int qs, Lq, ks, Lk, gq0, gk0; const float* mask; int mlen; };
  auto mha_stage = [&](int l, int mo, bool two, AttnRange r1, AttnRange r2) {
    const float *qw = Pp(l, mo, 0), *qb = Pp(l, mo, 1), *kw = Pp(l, mo, 2), *kb = Pp(l, mo, 3),
                *vw = Pp(l, mo, 4), *vb = Pp(l, mo, 5), *ow = Pp(l, mo, 6), *ob = Pp(l, mo, 7),
                *lg = Pp(l, mo, 8), *lb = Pp(l, mo, 9);
    const int M = NB * NTOK;
    gemm(cur, CAPE,    qw, qb, Qb, M, BD, BD, 0, NTOK, 0);  // (x + ape) @ qw + qb
    gemm(cur, CAPE,    kw, kb, Kb, M, BD, BD, 0, NTOK, 0);
    gemm(cur, nullptr, vw, vb, Vb, M, BD, BD, 0, NTOK, 0);
    auto att = [&](AttnRange r) {
      attention<<<dim3(r.Lq, NH, NB), 128, 0, stream>>>(
          Qb, Kb, Vb, G, r.mask, r.mlen, Cb, r.qs, r.Lq, r.ks, r.Lk, r.gq0, r.gk0);
    };
    att(r1);
    if (two) att(r2);
    gemm(Cb, nullptr, ow, ob, Y, M, BD, BD, 0, NTOK, 0);
    add_layernorm<<<M, 256, 0, stream>>>(cur, Y, lg, lb, nxt, 0, NTOK, NTOK);
    swapbuf();
  };

  auto ffn_run = [&](int l, int mo, int rowStart, int rowCount) {
    const float *w1 = Pp(l, mo, 0), *b1 = Pp(l, mo, 1), *w2 = Pp(l, mo, 2),
                *b2 = Pp(l, mo, 3), *lg = Pp(l, mo, 4), *lb = Pp(l, mo, 5);
    const int M = NB * rowCount;
    gemm(cur, nullptr, w1, b1, Hh, M, FD, BD, rowStart, rowCount, 2 /*gelu*/);
    gemm(Hh,  nullptr, w2, b2, Y,  M, BD, FD, rowStart, rowCount, 0);
    add_layernorm<<<M, 256, 0, stream>>>(cur, Y, lg, lb, nxt, rowStart, rowCount, NTOK);
  };

  for (int l = 0; l < 3; ++l) {
    // self attention: shared weights; visn att (mask=vmask, rpe=r2r), text att (rpe=t2t)
    mha_stage(l, 0, true,
              { NTXT, NVIS, NTXT, NVIS, NTXT, NTXT, vmask, NVIS },
              { 0,    NTXT, 0,    NTXT, 0,    0,    nullptr, 0 });
    ffn_run(l, 30, NTXT, NVIS);  // visn_self_ffn
    ffn_run(l, 36, 0,    NTXT);  // text_self_ffn
    swapbuf();
    // cross attention: v queries / t keys (rpe=r2t); t queries / v keys (mask, rpe=t2r)
    mha_stage(l, 10, true,
              { NTXT, NVIS, 0,    NTXT, NTXT, 0,    nullptr, 0 },
              { 0,    NTXT, NTXT, NVIS, 0,    NTXT, vmask, NVIS });
    ffn_run(l, 42, NTXT, NVIS);  // visn_cross_ffn
    ffn_run(l, 48, 0,    NTXT);  // text_cross_ffn
    swapbuf();
    // mix attention over all 280 tokens (mask=cat_mask, rpe=g), then text_self_ffn
    mha_stage(l, 20, false,
              { 0, NTOK, 0, NTOK, 0, 0, CM, NTOK },
              { 0, NTOK, 0, NTOK, 0, 0, CM, NTOK });
    ffn_run(l, 36, 0, NTOK);
    swapbuf();
  }

  write_out<<<NB * NTOK, 256, 0, stream>>>(cur, (float*)d_out);
}
